// LinearAttention_51900384805206
// MI455X (gfx1250) — matrix operand order repaired and checked by reading, not compiled, not run on hardware
//
#include <hip/hip_runtime.h>
#include <math.h>

// ---------------------------------------------------------------------------
// LinearAttention for MI455X (gfx1250, wave32, WMMA), v2.
//
//  P0 convert_bf    : w_qkv f32 -> bf16
//  P1 transpose_x   : x[b][c][n] f32 -> xT[b][n][c] bf16  (B operand layout)
//  K1 gemm_bf16     : qkv = w_qkv @ x ; epilogue: q -> qT[b][n][hd] bf16
//                     (pre-scaled), k/v -> kv[b][0..255][n] f32
//  K2 reduce_k      : per k-row max / 1/sum(exp)
//  K3 context_part  : partial ctx[b,h,d,e] (16 n-partitions, no atomics)
//  K4 ctx_reduce    : sum partials
//  K5 build_w2      : W2[b,o,hd] bf16 = sum_e w_out[o,h*32+e] ctx[b,h,d,e]
//  K6 gemm_bf16     : out[b] = W2[b] @ qT[b] + b_out  (f32 out)
//
// All f32->bf16 conversion happens in bandwidth-bound passes; the WMMA GEMM
// inner loop is pure b128 copies + fragment loads + v_wmma.
// ---------------------------------------------------------------------------

typedef float  v8f   __attribute__((ext_vector_type(8)));
typedef __bf16 v16bf __attribute__((ext_vector_type(16)));
typedef unsigned short u16;

#define HEADS 4
#define DHEAD 32
#define HIDDEN 128
#define CDIM 256
#define BATCH 16
#define NTOK 16384
#define QSCALE 0.17677669529663687f   // 32^-0.5

__device__ __forceinline__ unsigned f2bf(float f) {
    unsigned u = __float_as_uint(f);
    return (u + 0x7FFFu + ((u >> 16) & 1u)) >> 16;   // RNE to bf16 bits
}

union FragBF { v16bf v; uint4 q[2]; };

// ---------------------------------------------------------------------------
// bf16 WMMA GEMM.  A: [M][K] bf16 row-major (read direct from global, L2-hot)
//                  B: [N][K] bf16 n-major  (staged LDS, pure b128 copies)
// WG=256 (8 waves), tile M64 x N128, K-step 32.
// Waves: 4(M) x 2(N); each wave = 16x64 = four 16x16 f32 accs.
// mode 1: rows<128 -> qT bf16 (scaled), rows>=128 -> kv f32.
// mode 0: C f32 row-major + bias.
// ---------------------------------------------------------------------------
#define TM 64
#define TN 128
#define TK 32
#define BPITCH 40   // ushort pitch for BsT rows (80B: 16B aligned, odd banks)

__global__ __launch_bounds__(256)
void gemm_bf16(const u16* __restrict__ Abf, unsigned long long aBS,
               const u16* __restrict__ Bt,  unsigned long long bBS, int ldbk,
               int M, int K, int N, int mode, float scale,
               const float* __restrict__ bias,
               float* __restrict__ Cf, unsigned long long cBS,
               u16* __restrict__ qT, float* __restrict__ kv) {
    __shared__ u16 BsT[TN][BPITCH];

    const int tid  = threadIdx.x;
    const int wid  = tid >> 5;
    const int lane = tid & 31;
    const int half = lane >> 4;
    const int lrow = lane & 15;
    const int wm   = (wid >> 1) * 16;
    const int wn   = (wid & 1) * 64;

    const int n0g = blockIdx.x * TN;
    const int m0g = blockIdx.y * TM;
    const int bz  = blockIdx.z;
    const u16* A = Abf + (unsigned long long)bz * aBS;
    const u16* B = Bt  + (unsigned long long)bz * bBS;

    v8f acc0 = {}, acc1 = {}, acc2 = {}, acc3 = {};

    const int kb = half * 16;
    const int arow = m0g + wm + lrow;

    for (int k0 = 0; k0 < K; k0 += TK) {
        // ---- stage B tile: 128 rows x 32 k bf16 (8KB), pure b128 copies
        {
            int row = tid >> 1;
            int seg = (tid & 1) * 16;                 // ushort offset
            const uint4* src = (const uint4*)(B + (size_t)(n0g + row) * ldbk + k0 + seg);
            uint4* dst = (uint4*)&BsT[row][seg];
            dst[0] = src[0];
            dst[1] = src[1];
        }
        if (k0 + TK < K)
            __builtin_prefetch(B + (size_t)(n0g + (tid & 127)) * ldbk + k0 + TK, 0, 1);
        __syncthreads();

        // ---- A fragment: direct global, two contiguous 16B runs
        //  (K = 16*half + 0..15, contiguous, matching B fragments)
        FragBF af;
        af.q[0] = *(const uint4*)(A + (size_t)arow * K + k0 + kb);
        af.q[1] = *(const uint4*)(A + (size_t)arow * K + k0 + kb + 8);

        // ---- B fragments: K = 16*half + 0..15, contiguous 32B in BsT row
        #pragma unroll
        for (int s = 0; s < 4; ++s) {
            FragBF bf;
            const uint4* bp = (const uint4*)&BsT[wn + s * 16 + lrow][16 * half];
            bf.q[0] = bp[0];
            bf.q[1] = bp[1];
            v8f* pa = (s == 0) ? &acc0 : (s == 1) ? &acc1 : (s == 2) ? &acc2 : &acc3;
            *pa = __builtin_amdgcn_wmma_f32_16x16x32_bf16(
                      false, af.v, false, bf.v, (short)0, *pa, false, false);
        }
        __syncthreads();
    }

    // ---- epilogue (C layout: m = r + 8*half contiguous per lane, n = lane&15)
    const int mb = m0g + wm;                 // multiple of 16
    #pragma unroll
    for (int s = 0; s < 4; ++s) {
        v8f a = (s == 0) ? acc0 : (s == 1) ? acc1 : (s == 2) ? acc2 : acc3;
        int n = n0g + wn + s * 16 + lrow;
        if (mode == 1) {
            if (mb < HIDDEN) {
                // q rows: scaled bf16, transposed store (8 contiguous m = 16B)
                uint4 p;
                p.x = f2bf(a[0] * scale) | (f2bf(a[1] * scale) << 16);
                p.y = f2bf(a[2] * scale) | (f2bf(a[3] * scale) << 16);
                p.z = f2bf(a[4] * scale) | (f2bf(a[5] * scale) << 16);
                p.w = f2bf(a[6] * scale) | (f2bf(a[7] * scale) << 16);
                *(uint4*)(qT + ((size_t)bz * NTOK + n) * HIDDEN + mb + half * 8) = p;
            } else {
                #pragma unroll
                for (int r = 0; r < 8; ++r) {
                    int row = mb + half * 8 + r - HIDDEN;   // 0..255 (k then v)
                    kv[((size_t)bz * 256 + row) * NTOK + n] = a[r];
                }
            }
        } else {
            #pragma unroll
            for (int r = 0; r < 8; ++r) {
                int m = mb + half * 8 + r;
                Cf[(unsigned long long)bz * cBS + (size_t)m * N + n] = a[r] + bias[m];
            }
        }
    }
}

// ---------------------------------------------------------------------------
// P0: f32 -> bf16 flat convert (w_qkv)
// ---------------------------------------------------------------------------
__global__ __launch_bounds__(256)
void convert_bf(const float* __restrict__ src, u16* __restrict__ dst, int n) {
    int i = blockIdx.x * 256 + threadIdx.x;
    if (i < n) dst[i] = (u16)f2bf(src[i]);
}

// ---------------------------------------------------------------------------
// P1: x[b][c][n] f32 -> xT[b][n][c] bf16, 64x64 tiles via LDS
// ---------------------------------------------------------------------------
__global__ __launch_bounds__(256)
void transpose_x_bf(const float* __restrict__ x, u16* __restrict__ xT) {
    __shared__ u16 tl[64][68];
    int b = blockIdx.z, c0 = blockIdx.y * 64, n0 = blockIdx.x * 64;
    int tid = threadIdx.x;
    #pragma unroll
    for (int i = 0; i < 4; ++i) {
        int lin = tid + i * 256;                 // 1024 float4 of 64x64 tile
        int c = lin >> 4;
        int n4 = (lin & 15) * 4;
        float4 v = *(const float4*)(x + ((size_t)b * CDIM + c0 + c) * NTOK + n0 + n4);
        uint2 p;
        p.x = f2bf(v.x) | (f2bf(v.y) << 16);
        p.y = f2bf(v.z) | (f2bf(v.w) << 16);
        *(uint2*)&tl[c][n4] = p;
    }
    __syncthreads();
    int nl = tid >> 2;
    int cs = (tid & 3) * 16;
    unsigned pk[8];
    #pragma unroll
    for (int j = 0; j < 8; ++j)
        pk[j] = (unsigned)tl[cs + 2 * j][nl] | ((unsigned)tl[cs + 2 * j + 1][nl] << 16);
    u16* dst = xT + ((size_t)b * NTOK + n0 + nl) * CDIM + c0 + cs;
    uint4 q0; q0.x = pk[0]; q0.y = pk[1]; q0.z = pk[2]; q0.w = pk[3];
    uint4 q1; q1.x = pk[4]; q1.y = pk[5]; q1.z = pk[6]; q1.w = pk[7];
    *(uint4*)dst = q0;
    *(uint4*)(dst + 8) = q1;
}

// ---------------------------------------------------------------------------
// K2: per-row (b,hd) max and 1/sum(exp) over k rows (kv rows 0..127)
// ---------------------------------------------------------------------------
__global__ __launch_bounds__(256)
void reduce_k(const float* __restrict__ kv, float* __restrict__ rowmax,
              float* __restrict__ rowinv) {
    int row = blockIdx.x;                 // 0..2047
    int b = row >> 7, hd = row & 127;
    const float* p = kv + ((size_t)b * 256 + hd) * NTOK;
    __shared__ float sred[256];
    int tid = threadIdx.x;

    float m = -INFINITY;
    for (int i = tid; i < NTOK; i += 256) m = fmaxf(m, p[i]);
    sred[tid] = m; __syncthreads();
    for (int s = 128; s > 0; s >>= 1) {
        if (tid < s) sred[tid] = fmaxf(sred[tid], sred[tid + s]);
        __syncthreads();
    }
    float rm = sred[0]; __syncthreads();

    float sum = 0.f;
    for (int i = tid; i < NTOK; i += 256) sum += __expf(p[i] - rm);
    sred[tid] = sum; __syncthreads();
    for (int s = 128; s > 0; s >>= 1) {
        if (tid < s) sred[tid] += sred[tid + s];
        __syncthreads();
    }
    if (tid == 0) { rowmax[row] = rm; rowinv[row] = 1.0f / sred[0]; }
}

// ---------------------------------------------------------------------------
// K3: partial context over 1024-token slices (deterministic, no atomics)
// ---------------------------------------------------------------------------
__global__ __launch_bounds__(256)
void context_part(const float* __restrict__ kv, const float* __restrict__ rowmax,
                  const float* __restrict__ rowinv, float* __restrict__ ctxp) {
    int part = blockIdx.x, h = blockIdx.y, b = blockIdx.z;
    const float* kb_ = kv + ((size_t)b * 256 + h * DHEAD) * NTOK;
    const float* vb_ = kv + ((size_t)b * 256 + 128 + h * DHEAD) * NTOK;
    const float* rm = rowmax + b * 128 + h * DHEAD;
    const float* ri = rowinv + b * 128 + h * DHEAD;

    __shared__ float pc[32][129];
    __shared__ float vc[32][129];

    int tid = threadIdx.x;
    int e = tid & 31, dg = tid >> 5;
    float a0 = 0.f, a1 = 0.f, a2 = 0.f, a3 = 0.f;
    int n0 = part * 1024;

    for (int c0 = 0; c0 < 1024; c0 += 128) {
        #pragma unroll
        for (int i = 0; i < 4; ++i) {
            int lin = tid + i * 256;
            int r = lin >> 5;
            int c4 = (lin & 31) * 4;
            float m = rm[r], inv = ri[r];
            float4 kk = *(const float4*)(kb_ + (size_t)r * NTOK + n0 + c0 + c4);
            pc[r][c4 + 0] = __expf(kk.x - m) * inv;
            pc[r][c4 + 1] = __expf(kk.y - m) * inv;
            pc[r][c4 + 2] = __expf(kk.z - m) * inv;
            pc[r][c4 + 3] = __expf(kk.w - m) * inv;
            float4 vv = *(const float4*)(vb_ + (size_t)r * NTOK + n0 + c0 + c4);
            vc[r][c4 + 0] = vv.x; vc[r][c4 + 1] = vv.y;
            vc[r][c4 + 2] = vv.z; vc[r][c4 + 3] = vv.w;
        }
        __syncthreads();
        for (int nn = 0; nn < 128; ++nn) {
            float vv = vc[e][nn];
            a0 += pc[dg + 0][nn] * vv;
            a1 += pc[dg + 8][nn] * vv;
            a2 += pc[dg + 16][nn] * vv;
            a3 += pc[dg + 24][nn] * vv;
        }
        __syncthreads();
    }
    float* cb = ctxp + ((size_t)part * BATCH * HEADS + (size_t)b * HEADS + h) * 1024;
    cb[(dg + 0)  * 32 + e] = a0;
    cb[(dg + 8)  * 32 + e] = a1;
    cb[(dg + 16) * 32 + e] = a2;
    cb[(dg + 24) * 32 + e] = a3;
}

// K4: sum 16 partials
__global__ __launch_bounds__(256)
void ctx_reduce(const float* __restrict__ ctxp, float* __restrict__ ctx) {
    int i = blockIdx.x * 256 + threadIdx.x;
    float s = 0.f;
    #pragma unroll
    for (int p = 0; p < 16; ++p) s += ctxp[(size_t)p * 65536 + i];
    ctx[i] = s;
}

// K5: W2[b][o][h*32+d] = sum_e w_out[o][h*32+e] * ctx[b][h][d][e]  -> bf16
__global__ __launch_bounds__(256)
void build_w2(const float* __restrict__ w_out, const float* __restrict__ ctx,
              u16* __restrict__ W2) {
    int b = blockIdx.x;
    int o = blockIdx.y * 2 + (threadIdx.x >> 7);
    int hd = threadIdx.x & 127;
    int h = hd >> 5, d = hd & 31;
    const float* c = ctx + (((size_t)b * HEADS + h) * 32 + d) * 32;
    const float* w = w_out + (size_t)o * HIDDEN + h * 32;
    float s = 0.f;
    #pragma unroll
    for (int e = 0; e < 32; ++e) s += w[e] * c[e];
    W2[((size_t)b * CDIM + o) * HIDDEN + hd] = (u16)f2bf(s);
}

// ---------------------------------------------------------------------------
extern "C" void kernel_launch(void* const* d_in, const int* in_sizes, int n_in,
                              void* d_out, int out_size, void* d_ws, size_t ws_size,
                              hipStream_t stream) {
    const float* x     = (const float*)d_in[0];
    const float* w_qkv = (const float*)d_in[1];
    const float* w_out = (const float*)d_in[2];
    const float* b_out = (const float*)d_in[3];
    float* out = (float*)d_out;

    // workspace carve (bf16 region first, 16B-aligned throughout)
    u16* xT  = (u16*)d_ws;                                   // 67,108,864
    u16* qT  = xT  + (size_t)BATCH * NTOK * CDIM;            // 33,554,432
    u16* wbf = qT  + (size_t)BATCH * NTOK * HIDDEN;          // 98,304
    u16* W2b = wbf + 384 * CDIM;                             // 524,288
    float* kv     = (float*)(W2b + (size_t)BATCH * CDIM * HIDDEN);
    float* rowmax = kv + (size_t)BATCH * 256 * NTOK;         // 2048
    float* rowinv = rowmax + 2048;                           // 2048
    float* ctxp   = rowinv + 2048;                           // 1,048,576
    float* ctx    = ctxp + (size_t)16 * 65536;               // 65,536

    dim3 blk(256);

    // P0/P1: bf16 operand prep
    convert_bf<<<dim3((384 * CDIM) / 256), blk, 0, stream>>>(w_qkv, wbf, 384 * CDIM);
    transpose_x_bf<<<dim3(NTOK / 64, CDIM / 64, BATCH), blk, 0, stream>>>(x, xT);

    // K1: qkv GEMM (M=384,K=256,N=16384); q->qT bf16 scaled, k/v->kv f32
    gemm_bf16<<<dim3(NTOK / TN, 384 / TM, BATCH), blk, 0, stream>>>(
        wbf, 0ull,
        xT, (unsigned long long)NTOK * CDIM, CDIM,
        384, CDIM, NTOK, /*mode=*/1, QSCALE,
        nullptr, nullptr, 0ull, qT, kv);

    // K2..K5: softmax stats, context, fold with w_out
    reduce_k<<<dim3(2048), blk, 0, stream>>>(kv, rowmax, rowinv);
    context_part<<<dim3(16, HEADS, BATCH), blk, 0, stream>>>(kv, rowmax, rowinv, ctxp);
    ctx_reduce<<<dim3(65536 / 256), blk, 0, stream>>>(ctxp, ctx);
    build_w2<<<dim3(BATCH, CDIM / 2), blk, 0, stream>>>(w_out, ctx, W2b);

    // K6: out[b] = W2[b] @ qT[b] + b_out  (M=256,K=128,N=16384)
    gemm_bf16<<<dim3(NTOK / TN, CDIM / TM, BATCH), blk, 0, stream>>>(
        W2b, (unsigned long long)CDIM * HIDDEN,
        qT, (unsigned long long)NTOK * HIDDEN, HIDDEN,
        CDIM, HIDDEN, NTOK, /*mode=*/0, 1.0f,
        b_out, out, (unsigned long long)CDIM * NTOK, nullptr, nullptr);
}
